// SparseLocalAttention_40948218200572
// MI455X (gfx1250) — matrix operand order repaired and checked by reading, not compiled, not run on hardware
//
#include <hip/hip_runtime.h>
#include <hip/hip_bf16.h>

// ---------------------------------------------------------------------------
// Sliding-window causal attention on MI455X (gfx1250, wave32, WMMA).
// B=128 T=256 C=384 H=6 hd=64 WINDOW=64.
// fp32 inputs -> bf16 operands + f32 WMMA accumulate (v_wmma_f32_16x16x32_bf16)
// - Fragments: pairs of ds_load_b128 via layout-aware LDS staging.
// - GEMMs: register-pipelined (global loads of tile k+1 overlap WMMAs of tile k).
// - Transposed staging (B, V) done as 4x4 / 4x8 register-block transposes:
//   b128 global loads + b64 LDS stores (no scalar b16 scatter).
// ---------------------------------------------------------------------------

typedef __attribute__((ext_vector_type(16))) __bf16 bf16x16;
typedef __attribute__((ext_vector_type(8)))  __bf16 bf16x8;
typedef __attribute__((ext_vector_type(4)))  __bf16 bf16x4;
typedef __attribute__((ext_vector_type(8)))  float  floatx8;
typedef __attribute__((ext_vector_type(4)))  float  floatx4;

#define B_    128
#define T_    256
#define C_    384
#define H_    6
#define HD_   64
#define N3C_  1152

__device__ __forceinline__ floatx8 zero8() {
  floatx8 z;
#pragma unroll
  for (int i = 0; i < 8; ++i) z[i] = 0.0f;
  return z;
}

__device__ __forceinline__ bf16x8 zero_bf8() {
  bf16x8 z;
#pragma unroll
  for (int i = 0; i < 8; ++i) z[i] = (__bf16)0.0f;
  return z;
}

__device__ __forceinline__ floatx8 wmma_bf16(bf16x16 a, bf16x16 b, floatx8 c) {
  return __builtin_amdgcn_wmma_f32_16x16x32_bf16(false, a, false, b, (short)0, c,
                                                 false, false);
}

// A fragment, 16x32 bf16 tile, row-major LDS (ld mult of 8 elems).
// a[0..7] = row[kb..kb+8), a[8..15] = row[kb+16..kb+24), kb = (lane>=16)?8:0.
__device__ __forceinline__ bf16x16 load_frag_a(const __bf16* base, int ld, int lane) {
  const int row = lane & 15;
  const int kb  = (lane >> 4) << 3;
  const __bf16* p = base + row * ld + kb;
  union { bf16x16 v; bf16x8 h[2]; } u;
  u.h[0] = *(const bf16x8*)p;
  u.h[1] = *(const bf16x8*)(p + 16);
  return u.v;
}

// B fragment, 32x16 tile stored TRANSPOSED in LDS: src[n][k], ld mult of 8.
// b[0..7] = col[kb..kb+8), b[8..15] = col[kb+16..kb+24), kb = (lane>=16)?8:0.
__device__ __forceinline__ bf16x16 load_frag_bt(const __bf16* base, int ld, int lane) {
  const int col = lane & 15;
  const int kb  = (lane >> 4) << 3;
  const __bf16* p = base + col * ld + kb;
  union { bf16x16 v; bf16x8 h[2]; } u;
  u.h[0] = *(const bf16x8*)p;
  u.h[1] = *(const bf16x8*)(p + 16);
  return u.v;
}

__device__ __forceinline__ bf16x4 cvt4(floatx4 f) {
  bf16x4 b4;
#pragma unroll
  for (int i = 0; i < 4; ++i) b4[i] = (__bf16)f[i];
  return b4;
}

// ---------------------------------------------------------------------------
// Kernel 1: qkv = x @ W_attn + b_attn   -> q,k,v as bf16 [B,H,T,64]
// grid (256, 9), block 256 (8 waves); 128x128 tile, BK=32, reg-pipelined.
// ---------------------------------------------------------------------------
__global__ __launch_bounds__(256) void qkv_gemm_kernel(
    const float* __restrict__ x, const float* __restrict__ W,
    const float* __restrict__ bias, __bf16* __restrict__ q,
    __bf16* __restrict__ k, __bf16* __restrict__ v) {
  constexpr int BM = 128, BN = 128, BK = 32;
  constexpr int LDA = BK + 8;   // 80B rows, 16B aligned
  constexpr int LDB = BK + 8;   // Bt[n][k]
  __shared__ __align__(16) __bf16 As[BM * LDA];
  __shared__ __align__(16) __bf16 Bt[BN * LDB];

  const int tid  = threadIdx.x;
  const int lane = tid & 31;
  const int wave = tid >> 5;
  const int wm   = wave & 3;
  const int wn   = wave >> 2;
  const int M0   = blockIdx.x * BM;
  const int N0   = blockIdx.y * BN;

  // A staging coords: 1024 float4 (128 rows x 8 per row), 4 per thread
  int ar[4], ac[4];
#pragma unroll
  for (int it = 0; it < 4; ++it) {
    int ia = tid + it * 256;
    ar[it] = ia >> 3;  ac[it] = (ia & 7) * 4;
  }
  // B staging: each thread owns a 4(k) x 4(n) block of the 32x128 tile
  const int kblk = tid >> 5;      // 0..7  -> k = kblk*4 + e
  const int nblk = tid & 31;      // 0..31 -> n = nblk*4 + e2

  floatx8 acc[2][4];
#pragma unroll
  for (int i = 0; i < 2; ++i)
#pragma unroll
    for (int j = 0; j < 4; ++j) acc[i][j] = zero8();

  floatx4 ra[4], rb[4];
  // prologue: load tile kk=0
#pragma unroll
  for (int it = 0; it < 4; ++it)
    ra[it] = *(const floatx4*)&x[(size_t)(M0 + ar[it]) * C_ + ac[it]];
#pragma unroll
  for (int e = 0; e < 4; ++e)
    rb[e] = *(const floatx4*)&W[(size_t)(kblk * 4 + e) * N3C_ + N0 + nblk * 4];

  for (int kk = 0; kk < C_; kk += BK) {
    // store current tile regs -> LDS (bf16); B via register transpose
#pragma unroll
    for (int it = 0; it < 4; ++it)
      *(bf16x4*)&As[ar[it] * LDA + ac[it]] = cvt4(ra[it]);
#pragma unroll
    for (int e2 = 0; e2 < 4; ++e2) {
      bf16x4 t;
      t[0] = (__bf16)rb[0][e2]; t[1] = (__bf16)rb[1][e2];
      t[2] = (__bf16)rb[2][e2]; t[3] = (__bf16)rb[3][e2];
      *(bf16x4*)&Bt[(nblk * 4 + e2) * LDB + kblk * 4] = t;
    }
    __syncthreads();

    // issue next tile's global loads (overlap with WMMAs below)
    if (kk + BK < C_) {
#pragma unroll
      for (int it = 0; it < 4; ++it)
        ra[it] = *(const floatx4*)&x[(size_t)(M0 + ar[it]) * C_ + kk + BK + ac[it]];
#pragma unroll
      for (int e = 0; e < 4; ++e)
        rb[e] = *(const floatx4*)&W[(size_t)(kk + BK + kblk * 4 + e) * N3C_ +
                                    N0 + nblk * 4];
    }

    bf16x16 af[2], bfm[4];
#pragma unroll
    for (int i = 0; i < 2; ++i)
      af[i] = load_frag_a(&As[(wm * 32 + i * 16) * LDA], LDA, lane);
#pragma unroll
    for (int j = 0; j < 4; ++j)
      bfm[j] = load_frag_bt(&Bt[(wn * 64 + j * 16) * LDB], LDB, lane);
#pragma unroll
    for (int i = 0; i < 2; ++i)
#pragma unroll
      for (int j = 0; j < 4; ++j) acc[i][j] = wmma_bf16(af[i], bfm[j], acc[i][j]);
    __syncthreads();
  }

  // epilogue: +bias, scatter into q/k/v [B,H,T,64] bf16
#pragma unroll
  for (int i = 0; i < 2; ++i) {
#pragma unroll
    for (int j = 0; j < 4; ++j) {
      int n      = N0 + wn * 64 + j * 16 + (lane & 15);
      int mbase  = M0 + wm * 32 + i * 16 + ((lane >> 4) << 3);
      float bb   = bias[n];
      int region = n / C_;          // 0=q 1=k 2=v
      int c      = n - region * C_;
      int h      = c >> 6;
      int d      = c & 63;
      __bf16* dst = (region == 0) ? q : ((region == 1) ? k : v);
#pragma unroll
      for (int r = 0; r < 8; ++r) {
        int m = mbase + r;
        int b = m >> 8;    // / T_
        int t = m & 255;
        dst[(((size_t)b * H_ + h) * T_ + t) * HD_ + d] = (__bf16)(acc[i][j][r] + bb);
      }
    }
  }
}

// ---------------------------------------------------------------------------
// Kernel 2: local causal attention, one block per (b,h, 64-query tile).
// Keys window: j in [q0-64, q0+64). Phased ~50KB LDS reuse:
//   phase1: Qs[64][72] @0, Ks[128][72] @9216
//   phase2: Ss f32 [64][128] @0,  Ps bf16 [64][136] @32768
//   phase3: Vt bf16 [64][136] @0 (from regs loaded up front), Ps kept.
// ---------------------------------------------------------------------------
__global__ __launch_bounds__(256) void attn_kernel(
    const __bf16* __restrict__ q, const __bf16* __restrict__ k,
    const __bf16* __restrict__ v, __bf16* __restrict__ y) {
  constexpr int LDQ = HD_ + 8;   // 72
  constexpr int LDP = 128 + 8;   // 136
  __shared__ __align__(16) char smem[50176];
  __bf16* Qs = (__bf16*)smem;                // [64][72]
  __bf16* Ks = (__bf16*)(smem + 9216);       // [128][72]
  float*  Ss = (float*)smem;                 // [64][128]
  __bf16* Ps = (__bf16*)(smem + 32768);      // [64][136]
  __bf16* Vt = (__bf16*)smem;                // [64][136]

  const int tid  = threadIdx.x;
  const int lane = tid & 31;
  const int wave = tid >> 5;
  const int bh   = blockIdx.x >> 2;       // b*H + h
  const int qb   = blockIdx.x & 3;
  const int q0   = qb << 6;
  const int j0   = q0 - 64;

  const __bf16* qp = q + ((size_t)bh * T_ + q0) * HD_;
  const __bf16* kp = k + (size_t)bh * T_ * HD_;
  const __bf16* vp = v + (size_t)bh * T_ * HD_;

  // V staging: each thread owns 4(j) x 8(d) of the 128x64 tile
  const int jb = tid >> 3;      // 0..31 -> j = jb*4 + e
  const int db = tid & 7;       // 0..7  -> d = db*8 .. +8

  // ---- issue ALL global loads up front (Q:2, K:4, V:4 b128 per thread) ----
  bf16x8 rq[2], rk[4], rv[4];
#pragma unroll
  for (int it = 0; it < 2; ++it) {
    int idx = tid + it * 256;            // < 512
    int r = idx >> 3, c8 = idx & 7;
    rq[it] = *(const bf16x8*)&qp[(size_t)r * HD_ + c8 * 8];
  }
#pragma unroll
  for (int it = 0; it < 4; ++it) {
    int idx = tid + it * 256;            // < 1024
    int r = idx >> 3, c8 = idx & 7;
    int j = j0 + r;
    rk[it] = (j >= 0) ? *(const bf16x8*)&kp[(size_t)j * HD_ + c8 * 8] : zero_bf8();
  }
#pragma unroll
  for (int e = 0; e < 4; ++e) {
    int j = j0 + jb * 4 + e;
    rv[e] = (j >= 0) ? *(const bf16x8*)&vp[(size_t)j * HD_ + db * 8] : zero_bf8();
  }

  // ---- stage Q and K ----
#pragma unroll
  for (int it = 0; it < 2; ++it) {
    int idx = tid + it * 256;
    int r = idx >> 3, c8 = idx & 7;
    *(bf16x8*)&Qs[r * LDQ + c8 * 8] = rq[it];
  }
#pragma unroll
  for (int it = 0; it < 4; ++it) {
    int idx = tid + it * 256;
    int r = idx >> 3, c8 = idx & 7;
    *(bf16x8*)&Ks[r * LDQ + c8 * 8] = rk[it];
  }
  __syncthreads();

  // S = Q K^T : wave w owns key-tile w (16 keys) over 4 query tiles
  floatx8 sacc[4];
#pragma unroll
  for (int i = 0; i < 4; ++i) sacc[i] = zero8();
#pragma unroll
  for (int kk = 0; kk < HD_; kk += 32) {
    bf16x16 bf = load_frag_bt(&Ks[(wave * 16) * LDQ + kk], LDQ, lane);
#pragma unroll
    for (int i = 0; i < 4; ++i) {
      bf16x16 af = load_frag_a(&Qs[(i * 16) * LDQ + kk], LDQ, lane);
      sacc[i] = wmma_bf16(af, bf, sacc[i]);
    }
  }
  __syncthreads();  // Qs/Ks reads complete before Ss overlays them

  // masked, scaled scores -> Ss
  const float scale = 0.125f;  // 1/sqrt(64)
#pragma unroll
  for (int i = 0; i < 4; ++i) {
    int rowb = i * 16 + ((lane >> 4) << 3);
    int jl   = wave * 16 + (lane & 15);
#pragma unroll
    for (int r = 0; r < 8; ++r) {
      int rr = rowb + r;
      // causal j<=i <=> jl <= rr+64 ; window i-j<=64 <=> jl >= rr ; j>=0
      bool ok = (jl >= rr) && (jl <= rr + 64) && (j0 + jl >= 0);
      Ss[rr * 128 + jl] = ok ? sacc[i][r] * scale : -1e30f;
    }
  }
  __syncthreads();

  // fp32 softmax, one thread per query row; exp stored back into Ss
  if (tid < 64) {
    float m = -1e30f;
    for (int c = 0; c < 128; ++c) m = fmaxf(m, Ss[tid * 128 + c]);
    float s = 0.0f;
    for (int c = 0; c < 128; ++c) {
      float e = __expf(Ss[tid * 128 + c] - m);
      s += e;
      Ss[tid * 128 + c] = e;
    }
    float inv = 1.0f / s;
    for (int c = 0; c < 128; ++c)
      Ps[tid * LDP + c] = (__bf16)(Ss[tid * 128 + c] * inv);
  }
  __syncthreads();

  // phase 3: register-transpose V over the dead S region: Vt[d][j]
#pragma unroll
  for (int e2 = 0; e2 < 8; ++e2) {
    bf16x4 t;
    t[0] = rv[0][e2]; t[1] = rv[1][e2];
    t[2] = rv[2][e2]; t[3] = rv[3][e2];
    *(bf16x4*)&Vt[(db * 8 + e2) * LDP + jb * 4] = t;
  }
  __syncthreads();

  // O = P V : wave -> (mt = wave>>1, nt = (wave&1)*2 + {0,1})
  const int mt  = wave >> 1;
  const int ntb = (wave & 1) << 1;
  floatx8 oacc[2];
  oacc[0] = zero8();
  oacc[1] = zero8();
#pragma unroll
  for (int kk = 0; kk < 128; kk += 32) {
    bf16x16 af = load_frag_a(&Ps[(mt * 16) * LDP + kk], LDP, lane);
#pragma unroll
    for (int j2 = 0; j2 < 2; ++j2) {
      bf16x16 bfv = load_frag_bt(&Vt[((ntb + j2) * 16) * LDP + kk], LDP, lane);
      oacc[j2] = wmma_bf16(af, bfv, oacc[j2]);
    }
  }

  // write y as bf16 [B*T, 384] row-major for the projection GEMM
  const int b = bh / H_;
  const int h = bh % H_;
#pragma unroll
  for (int j2 = 0; j2 < 2; ++j2) {
    int d0   = (ntb + j2) * 16 + (lane & 15);
    int row0 = mt * 16 + ((lane >> 4) << 3);
#pragma unroll
    for (int r = 0; r < 8; ++r) {
      int t = q0 + row0 + r;
      y[((size_t)b * T_ + t) * C_ + h * HD_ + d0] = (__bf16)oacc[j2][r];
    }
  }
}

// ---------------------------------------------------------------------------
// Kernel 3: out = y @ W_proj + b_proj  (fp32 output), grid (256, 3),
// reg-pipelined like kernel 1 (A already bf16).
// ---------------------------------------------------------------------------
__global__ __launch_bounds__(256) void proj_gemm_kernel(
    const __bf16* __restrict__ y, const float* __restrict__ W,
    const float* __restrict__ bias, float* __restrict__ out) {
  constexpr int BM = 128, BN = 128, BK = 32;
  constexpr int LDA = BK + 8;
  constexpr int LDB = BK + 8;
  __shared__ __align__(16) __bf16 As[BM * LDA];
  __shared__ __align__(16) __bf16 Bt[BN * LDB];

  const int tid  = threadIdx.x;
  const int lane = tid & 31;
  const int wave = tid >> 5;
  const int wm   = wave & 3;
  const int wn   = wave >> 2;
  const int M0   = blockIdx.x * BM;
  const int N0   = blockIdx.y * BN;

  // A: 512 bf16x8 (128 rows x 4), 2 per thread; B: 4x4 block per thread
  int ar[2], ac[2];
#pragma unroll
  for (int it = 0; it < 2; ++it) {
    int ia = tid + it * 256;
    ar[it] = ia >> 2;  ac[it] = (ia & 3) * 8;
  }
  const int kblk = tid >> 5;
  const int nblk = tid & 31;

  floatx8 acc[2][4];
#pragma unroll
  for (int i = 0; i < 2; ++i)
#pragma unroll
    for (int j = 0; j < 4; ++j) acc[i][j] = zero8();

  bf16x8 ra[2];
  floatx4 rb[4];
#pragma unroll
  for (int it = 0; it < 2; ++it)
    ra[it] = *(const bf16x8*)&y[(size_t)(M0 + ar[it]) * C_ + ac[it]];
#pragma unroll
  for (int e = 0; e < 4; ++e)
    rb[e] = *(const floatx4*)&W[(size_t)(kblk * 4 + e) * C_ + N0 + nblk * 4];

  for (int kk = 0; kk < C_; kk += BK) {
#pragma unroll
    for (int it = 0; it < 2; ++it)
      *(bf16x8*)&As[ar[it] * LDA + ac[it]] = ra[it];
#pragma unroll
    for (int e2 = 0; e2 < 4; ++e2) {
      bf16x4 t;
      t[0] = (__bf16)rb[0][e2]; t[1] = (__bf16)rb[1][e2];
      t[2] = (__bf16)rb[2][e2]; t[3] = (__bf16)rb[3][e2];
      *(bf16x4*)&Bt[(nblk * 4 + e2) * LDB + kblk * 4] = t;
    }
    __syncthreads();

    if (kk + BK < C_) {
#pragma unroll
      for (int it = 0; it < 2; ++it)
        ra[it] = *(const bf16x8*)&y[(size_t)(M0 + ar[it]) * C_ + kk + BK + ac[it]];
#pragma unroll
      for (int e = 0; e < 4; ++e)
        rb[e] = *(const floatx4*)&W[(size_t)(kk + BK + kblk * 4 + e) * C_ +
                                    N0 + nblk * 4];
    }

    bf16x16 af[2], bfm[4];
#pragma unroll
    for (int i = 0; i < 2; ++i)
      af[i] = load_frag_a(&As[(wm * 32 + i * 16) * LDA], LDA, lane);
#pragma unroll
    for (int j = 0; j < 4; ++j)
      bfm[j] = load_frag_bt(&Bt[(wn * 64 + j * 16) * LDB], LDB, lane);
#pragma unroll
    for (int i = 0; i < 2; ++i)
#pragma unroll
      for (int j = 0; j < 4; ++j) acc[i][j] = wmma_bf16(af[i], bfm[j], acc[i][j]);
    __syncthreads();
  }

#pragma unroll
  for (int i = 0; i < 2; ++i) {
#pragma unroll
    for (int j = 0; j < 4; ++j) {
      int n     = N0 + wn * 64 + j * 16 + (lane & 15);
      int mbase = M0 + wm * 32 + i * 16 + ((lane >> 4) << 3);
      float bb  = bias[n];
#pragma unroll
      for (int r = 0; r < 8; ++r) {
        int m = mbase + r;
        out[(size_t)m * C_ + n] = acc[i][j][r] + bb;
      }
    }
  }
}

// ---------------------------------------------------------------------------
extern "C" void kernel_launch(void* const* d_in, const int* in_sizes, int n_in,
                              void* d_out, int out_size, void* d_ws, size_t ws_size,
                              hipStream_t stream) {
  const float* x      = (const float*)d_in[0];
  const float* W_attn = (const float*)d_in[1];
  const float* b_attn = (const float*)d_in[2];
  const float* W_proj = (const float*)d_in[3];
  const float* b_proj = (const float*)d_in[4];
  float* out          = (float*)d_out;

  const size_t per_mat = (size_t)B_ * H_ * T_ * HD_;  // 12,582,912 bf16 elems
  __bf16* qw = (__bf16*)d_ws;
  __bf16* kw = qw + per_mat;
  __bf16* vw = kw + per_mat;
  __bf16* yw = vw + per_mat;  // [B*T, 384] bf16

  qkv_gemm_kernel<<<dim3(B_ * T_ / 128, N3C_ / 128), 256, 0, stream>>>(
      x, W_attn, b_attn, qw, kw, vw);

  attn_kernel<<<dim3(B_ * H_ * (T_ / 64)), 256, 0, stream>>>(qw, kw, vw, yw);

  proj_gemm_kernel<<<dim3(B_ * T_ / 128, C_ / 128), 256, 0, stream>>>(
      yw, W_proj, b_proj, out);
}
